// GNN_Layer_11321533792934
// MI455X (gfx1250) — matrix operand order repaired and checked by reading, not compiled, not run on hardware
//
#include <hip/hip_runtime.h>
#include <hip/hip_bf16.h>
#include <math.h>

// ---------------------------------------------------------------------------
// TransformerConv (H=4 heads, C=32, F=128) for MI455X (gfx1250, wave32).
//  - all 4 projections fused in one WMMA GEMM kernel (x staged once per tile)
//  - X tile staged LDS via Tensor Data Mover (padded stride via D# pad fields)
//  - edge softmax/aggregation via L2-resident gathers + float atomics
// ---------------------------------------------------------------------------

typedef _Float16 v16h __attribute__((ext_vector_type(16)));
typedef _Float16 v8h  __attribute__((ext_vector_type(8)));
typedef float    v8f  __attribute__((ext_vector_type(8)));
typedef unsigned int u32x4 __attribute__((ext_vector_type(4)));
typedef int          i32x4 __attribute__((ext_vector_type(4)));
typedef int          i32x8 __attribute__((ext_vector_type(8)));

constexpr int FDIM  = 128;   // in features == H*C out features
constexpr int HEADS = 4;
constexpr int CDIM  = 32;
constexpr int TILE_M = 32;       // rows per block
constexpr int LDS_STRIDE = 136;  // 128 f16 + 16B pad -> 272B row stride

#if __has_builtin(__builtin_amdgcn_tensor_load_to_lds) && __has_builtin(__builtin_amdgcn_s_wait_tensorcnt)
#define USE_TDM 1
#else
#define USE_TDM 0
#endif

// ---------------- pre-convert x and the four W matrices to f16 --------------
__global__ void convert_f16_kernel(const float* __restrict__ x,
                                   const float* __restrict__ Wq, const float* __restrict__ Wk,
                                   const float* __restrict__ Wv, const float* __restrict__ Ws,
                                   _Float16* __restrict__ xh, _Float16* __restrict__ wh,
                                   int nx)
{
    const int i = blockIdx.x * blockDim.x + threadIdx.x;
    if (i < nx) xh[i] = (_Float16)x[i];
    if (i < 4 * FDIM * FDIM) {
        const int m = i >> 14;          // 128*128 = 16384 per matrix
        const int r = i & 16383;
        const float* Wsel = (m == 0) ? Wq : (m == 1) ? Wk : (m == 2) ? Wv : Ws;
        wh[i] = (_Float16)Wsel[r];
    }
}

// ---------------- per-matrix 32x16 WMMA slab ---------------------------------
__device__ __forceinline__ void gemm_slab(const _Float16* __restrict__ wmat,
                                          const float* __restrict__ bvec,
                                          float* __restrict__ Y,
                                          const _Float16* Xs,
                                          int tile_m, int tile_n,
                                          int half, int l15, int n, bool full)
{
    v8f c0 = {}, c1 = {};
    #pragma unroll
    for (int k0 = 0; k0 < FDIM; k0 += 32) {
        // A fragments (16x32 f16): lanes 0-15: K={0..7,16..23}; 16-31: K={8..15,24..31}
        union { v16h v; v8h h[2]; } a0, a1, b;
        const int aoff = k0 + half * 8;
        a0.h[0] = *(const v8h*)&Xs[l15 * LDS_STRIDE + aoff];
        a0.h[1] = *(const v8h*)&Xs[l15 * LDS_STRIDE + aoff + 16];
        a1.h[0] = *(const v8h*)&Xs[(16 + l15) * LDS_STRIDE + aoff];
        a1.h[1] = *(const v8h*)&Xs[(16 + l15) * LDS_STRIDE + aoff + 16];
        // B fragment (32x16 f16): N=l15, same K order as A; B[k][j]=W[j][k]
        const _Float16* wp = wmat + (tile_n + l15) * FDIM + k0 + half * 8;
        b.h[0] = *(const v8h*)wp;
        b.h[1] = *(const v8h*)(wp + 16);
        c0 = __builtin_amdgcn_wmma_f32_16x16x32_f16(false, a0.v, false, b.v,
                                                    (short)0, c0, false, false);
        c1 = __builtin_amdgcn_wmma_f32_16x16x32_f16(false, a1.v, false, b.v,
                                                    (short)0, c1, false, false);
    }
    const int   col  = tile_n + l15;
    const float bias = bvec[col];
    if (full) {
        #pragma unroll
        for (int r = 0; r < 8; ++r) {
            Y[(tile_m + r + half * 8) * FDIM + col]      = c0[r] + bias;
            Y[(tile_m + 16 + r + half * 8) * FDIM + col] = c1[r] + bias;
        }
    } else {
        #pragma unroll
        for (int r = 0; r < 8; ++r) {
            const int m0 = tile_m + r + half * 8;
            const int m1 = m0 + 16;
            if (m0 < n) Y[m0 * FDIM + col] = c0[r] + bias;
            if (m1 < n) Y[m1 * FDIM + col] = c1[r] + bias;
        }
    }
}

// ---------------- fused GEMM: Q,K,V,S = x @ W*^T + b* ------------------------
__global__ __launch_bounds__(256)
void proj_gemm_kernel(const _Float16* __restrict__ xh,
                      const _Float16* __restrict__ wh,   // [4][128][128] f16
                      const float* __restrict__ bq, const float* __restrict__ bk,
                      const float* __restrict__ bv, const float* __restrict__ bs,
                      float* __restrict__ Q, float* __restrict__ Kf,
                      float* __restrict__ V, float* __restrict__ S,
                      int n)
{
    __shared__ __align__(16) _Float16 Xs[TILE_M * LDS_STRIDE];

    const int tile_m = blockIdx.x * TILE_M;

#if USE_TDM
    // ---- Tensor Data Mover: 2D tile 128 x 32 (f16), padded 16B per 256B row
    if (threadIdx.x < 32) {       // one wave issues the DMA (TDM ignores EXEC)
        typedef __attribute__((address_space(3))) _Float16* lds_fp;
        const unsigned lds_off = (unsigned)(unsigned long long)(lds_fp)Xs;
        const unsigned long long gaddr =
            (unsigned long long)(const void*)(xh + (size_t)tile_m * FDIM);
        const unsigned rows = (unsigned)((n - tile_m) < TILE_M ? (n - tile_m) : TILE_M);

        u32x4 g0;
        g0[0] = 1u;                                   // count=1, user mode, no gather
        g0[1] = lds_off;                              // lds_addr (bytes)
        g0[2] = (unsigned)(gaddr & 0xFFFFFFFFull);    // global_addr[31:0]
        g0[3] = (unsigned)((gaddr >> 32) & 0x1FFFFFFull) | (2u << 30);  // addr[56:32] | type=2

        i32x8 g1;
        g1[0] = (int)((1u << 16)      // data_size = 2 bytes
                    | (1u << 20)      // pad_enable
                    | (5u << 22)      // pad_interval: 64 DWORDs (= one 256B row)
                    | (3u << 25));    // pad_amount: 4 DWORDs (= 16B)
        g1[1] = (int)(128u << 16);    // tensor_dim0 = 128 (low 16 bits)
        g1[2] = (int)(rows << 16);    // tensor_dim1 low 16 (OOB rows read as 0)
        g1[3] = (int)(128u << 16);    // tile_dim0 = 128
        g1[4] = (int)TILE_M;          // tile_dim1 = 32
        g1[5] = 128;                  // tensor_dim0_stride = 128 elements
        g1[6] = 0;
        g1[7] = 0;

        i32x4 g2 = {0, 0, 0, 0};
        i32x4 g3 = {0, 0, 0, 0};
#if defined(__clang_major__) && __clang_major__ >= 23
        i32x8 g4 = {0, 0, 0, 0, 0, 0, 0, 0};
        __builtin_amdgcn_tensor_load_to_lds(g0, g1, g2, g3, g4, 0);
#else
        __builtin_amdgcn_tensor_load_to_lds(g0, g1, g2, g3, 0);
#endif
        __builtin_amdgcn_s_wait_tensorcnt(0);
    }
#else
    // ---- fallback: vector-copy staging (f16, padded stride)
    {
        const int t = threadIdx.x;
        #pragma unroll
        for (int i = 0; i < 2; ++i) {
            const int c   = t * 2 + i;        // 512 16B-chunks
            const int row = c >> 4;
            const int cc  = (c & 15) * 8;
            int gr = tile_m + row; if (gr >= n) gr = n - 1;
            *(v8h*)&Xs[row * LDS_STRIDE + cc] = *(const v8h*)(xh + gr * FDIM + cc);
        }
    }
#endif
    __syncthreads();

    const int wave   = threadIdx.x >> 5;
    const int lane   = threadIdx.x & 31;
    const int tile_n = wave * 16;            // 8 waves cover the 128 output cols
    const int half   = lane >> 4;
    const int l15    = lane & 31 & 15;
    const bool full  = (tile_m + TILE_M) <= n;

    gemm_slab(wh + 0 * FDIM * FDIM, bq, Q,  Xs, tile_m, tile_n, half, l15, n, full);
    gemm_slab(wh + 1 * FDIM * FDIM, bk, Kf, Xs, tile_m, tile_n, half, l15, n, full);
    gemm_slab(wh + 2 * FDIM * FDIM, bv, V,  Xs, tile_m, tile_n, half, l15, n, full);
    gemm_slab(wh + 3 * FDIM * FDIM, bs, S,  Xs, tile_m, tile_n, half, l15, n, full);
}

// ---------------- helpers: monotonic uint encoding for float atomic-max -----
__device__ __forceinline__ unsigned enc_f(float f) {
    unsigned u = __float_as_uint(f);
    return (u & 0x80000000u) ? ~u : (u | 0x80000000u);
}
__device__ __forceinline__ float dec_f(unsigned e) {
    return __uint_as_float((e & 0x80000000u) ? (e & 0x7FFFFFFFu) : ~e);
}

__global__ void init_kernel(float* __restrict__ out, unsigned* __restrict__ smax,
                            float* __restrict__ denom, int n_out, int n_nh)
{
    const int i = blockIdx.x * blockDim.x + threadIdx.x;
    if (i < n_out) out[i] = 0.0f;
    if (i < n_nh)  { smax[i] = 0u; denom[i] = 0.0f; }
}

// ---------------- edge attention logits + segment max (one wave per edge) ---
__global__ __launch_bounds__(256)
void edge_score_kernel(const int* __restrict__ ei, const float* __restrict__ Q,
                       const float* __restrict__ Kf, float* __restrict__ score,
                       unsigned* __restrict__ smax, int E)
{
    const int wave = threadIdx.x >> 5;
    const int lane = threadIdx.x & 31;
    const int e = blockIdx.x * 8 + wave;
    if (e >= E) return;
    const int src = ei[e];
    const int dst = ei[E + e];
    const float* qv = Q  + dst * FDIM;
    const float* kv = Kf + src * FDIM;
    float p0 = qv[0*CDIM + lane] * kv[0*CDIM + lane];
    float p1 = qv[1*CDIM + lane] * kv[1*CDIM + lane];
    float p2 = qv[2*CDIM + lane] * kv[2*CDIM + lane];
    float p3 = qv[3*CDIM + lane] * kv[3*CDIM + lane];
    #pragma unroll
    for (int off = 16; off > 0; off >>= 1) {   // wave32 butterfly reduction
        p0 += __shfl_xor(p0, off, 32);
        p1 += __shfl_xor(p1, off, 32);
        p2 += __shfl_xor(p2, off, 32);
        p3 += __shfl_xor(p3, off, 32);
    }
    if (lane < HEADS) {                         // lanes 0-3 handle one head each
        const float inv_sqrt_c = 0.17677669529663687f;   // 1/sqrt(32)
        float s = (lane == 0) ? p0 : (lane == 1) ? p1 : (lane == 2) ? p2 : p3;
        s *= inv_sqrt_c;
        score[e * HEADS + lane] = s;
        atomicMax(&smax[dst * HEADS + lane], enc_f(s));
    }
}

// ---------------- exp(score - max) and segment-sum denominator --------------
__global__ void edge_exp_kernel(const int* __restrict__ ei, float* __restrict__ score,
                                const unsigned* __restrict__ smax,
                                float* __restrict__ denom, int E)
{
    const int i = blockIdx.x * blockDim.x + threadIdx.x;  // over E*HEADS
    if (i >= E * HEADS) return;
    const int e = i >> 2;
    const int h = i & 3;
    const int dst = ei[E + e];
    const float m = dec_f(smax[dst * HEADS + h]);
    const float a = __expf(score[i] - m);
    score[i] = a;
    atomicAdd(&denom[dst * HEADS + h], a);
}

// ---------------- scatter-aggregate alpha * v[src] into out[dst] ------------
__global__ __launch_bounds__(256)
void edge_agg_kernel(const int* __restrict__ ei, const float* __restrict__ score,
                     const float* __restrict__ denom, const float* __restrict__ V,
                     float* __restrict__ out, int E)
{
    const int wave = threadIdx.x >> 5;
    const int lane = threadIdx.x & 31;
    const int e = blockIdx.x * 8 + wave;
    if (e >= E) return;
    const int src = ei[e];
    const int dst = ei[E + e];
    #pragma unroll
    for (int h = 0; h < HEADS; ++h) {
        const float alpha = score[e * HEADS + h] / denom[dst * HEADS + h];
        const float val   = alpha * V[src * FDIM + h * CDIM + lane];
        atomicAdd(&out[dst * FDIM + h * CDIM + lane], val);
    }
}

// ---------------- skip connection + ReLU ------------------------------------
__global__ void finish_kernel(float* __restrict__ out, const float* __restrict__ S,
                              int total)
{
    const int i = blockIdx.x * blockDim.x + threadIdx.x;
    if (i >= total) return;
    const float v = out[i] + S[i];
    out[i] = v > 0.0f ? v : 0.0f;
}

// ---------------------------------------------------------------------------
extern "C" void kernel_launch(void* const* d_in, const int* in_sizes, int n_in,
                              void* d_out, int out_size, void* d_ws, size_t ws_size,
                              hipStream_t stream)
{
    const float* x  = (const float*)d_in[0];
    const int*   ei = (const int*)  d_in[1];
    const float* Wq = (const float*)d_in[2]; const float* bq = (const float*)d_in[3];
    const float* Wk = (const float*)d_in[4]; const float* bk = (const float*)d_in[5];
    const float* Wv = (const float*)d_in[6]; const float* bv = (const float*)d_in[7];
    const float* Ws = (const float*)d_in[8]; const float* bs = (const float*)d_in[9];
    float* out = (float*)d_out;

    const int n = in_sizes[0] / FDIM;   // 50000
    const int E = in_sizes[1] / 2;      // 800000

    // workspace carve-up (256B aligned)
    char*  ws  = (char*)d_ws;
    size_t off = 0;
    auto carve = [&](size_t bytes) -> char* {
        char* p = ws + off;
        off = (off + bytes + 255) & ~(size_t)255;
        return p;
    };
    float*     Q     = (float*)    carve((size_t)n * FDIM  * sizeof(float));
    float*     Kf    = (float*)    carve((size_t)n * FDIM  * sizeof(float));
    float*     V     = (float*)    carve((size_t)n * FDIM  * sizeof(float));
    float*     S     = (float*)    carve((size_t)n * FDIM  * sizeof(float));
    float*     score = (float*)    carve((size_t)E * HEADS * sizeof(float));
    unsigned*  smax  = (unsigned*) carve((size_t)n * HEADS * sizeof(unsigned));
    float*     denom = (float*)    carve((size_t)n * HEADS * sizeof(float));
    _Float16*  xh    = (_Float16*) carve((size_t)n * FDIM  * sizeof(_Float16));
    _Float16*  wh    = (_Float16*) carve((size_t)4 * FDIM * FDIM * sizeof(_Float16));

    const int nOut = n * FDIM;
    const int nNH  = n * HEADS;

    // 0) one streaming pass: f32 -> f16 copies of x and the four W matrices
    convert_f16_kernel<<<(nOut + 255) / 256, 256, 0, stream>>>(x, Wq, Wk, Wv, Ws,
                                                               xh, wh, nOut);

    // 1) fused projections on the WMMA tensor path (TDM-staged X tiles)
    proj_gemm_kernel<<<(n + TILE_M - 1) / TILE_M, 256, 0, stream>>>(
        xh, wh, bq, bk, bv, bs, Q, Kf, V, S, n);

    // 2) init accumulators
    init_kernel<<<(nOut + 255) / 256, 256, 0, stream>>>(out, smax, denom, nOut, nNH);

    // 3) logits + segment max
    edge_score_kernel<<<(E + 7) / 8, 256, 0, stream>>>(ei, Q, Kf, score, smax, E);

    // 4) exp + segment sum
    const int EH = E * HEADS;
    edge_exp_kernel<<<(EH + 255) / 256, 256, 0, stream>>>(ei, score, smax, denom, E);

    // 5) weighted scatter-aggregate
    edge_agg_kernel<<<(E + 7) / 8, 256, 0, stream>>>(ei, score, denom, V, out, E);

    // 6) skip + ReLU
    finish_kernel<<<(nOut + 255) / 256, 256, 0, stream>>>(out, S, nOut);
}
